// FastWeightGluMLPMultihead_31430570672197
// MI455X (gfx1250) — compile-verified, matrix operand order checked by reading, not run on hardware
//
#include <hip/hip_runtime.h>
#include <hip/hip_bf16.h>
#include <math.h>
#include <stdint.h>

typedef __bf16 bf16;
typedef __attribute__((ext_vector_type(16))) __bf16 v16bf;
typedef __attribute__((ext_vector_type(8)))  float   v8f;

#define A_NSC 3.4445f
#define B_NSC (-4.775f)
#define C_NSC 2.0315f
#define BASE_LR_INV_F (-4.6000701f)   // 0.01 + log(-expm1(-0.01))

// ---------------------------------------------------------------- helpers
__device__ __forceinline__ float sigm(float x){ return 1.0f/(1.0f+__expf(-x)); }
__device__ __forceinline__ float siluf(float x){ return x*sigm(x); }
__device__ __forceinline__ float silu_bp(float dy, float x){
  float s = sigm(x); return dy*s*(1.0f + x*(1.0f-s));
}

__device__ __forceinline__ v8f wmma_bf16(v16bf a, v16bf b, v8f c){
  // v_wmma_f32_16x16x32_bf16  (CDNA5, wave32)
  return __builtin_amdgcn_wmma_f32_16x16x32_bf16(false, a, false, b, (short)0, c, false, false);
}

// ------------------------------------------------ async global->LDS (CDNA5)
#if __has_builtin(__builtin_amdgcn_global_load_async_to_lds_b128)
#define ASYNC_LDS 1
#else
#define ASYNC_LDS 0
#endif

typedef int v4i_t __attribute__((vector_size(16)));
typedef __attribute__((address_space(1))) v4i_t v4i_g;   // global (device) int4
typedef __attribute__((address_space(3))) v4i_t v4i_l;   // LDS int4

__device__ __forceinline__ void cp16_async(const bf16* g, bf16* l){
#if ASYNC_LDS
  __builtin_amdgcn_global_load_async_to_lds_b128(
      (v4i_g*)(uintptr_t)g, (v4i_l*)(unsigned)(uintptr_t)l, 0, 0);
#else
  *(uint4*)l = *(const uint4*)g;
#endif
}
__device__ __forceinline__ void async_wait0(){
#if ASYNC_LDS
#if __has_builtin(__builtin_amdgcn_s_wait_asynccnt)
  __builtin_amdgcn_s_wait_asynccnt(0);
#else
  asm volatile("s_wait_asynccnt 0x0" ::: "memory");
#endif
#endif
}

// CDNA5 16-bit A fragment K-map (CDNA5 ISA §7.12.2):
// lanes 0-15: elems 0..7 -> K0..7,  elems 8..15 -> K16..23
// lanes16-31: elems 0..7 -> K8..15, elems 8..15 -> K24..31
__device__ __forceinline__ int kmap(int e, int hi){ return e + ((e>>3)<<3) + (hi<<3); }

__device__ __forceinline__ v16bf frag_A(const bf16* src, int ld){
  int lane = threadIdx.x & 31; int m = lane & 15; int hi = lane >> 4;
  v16bf f;
#pragma unroll
  for(int e=0;e<16;e++) f[e] = src[m*ld + kmap(e,hi)];
  return f;
}
__device__ __forceinline__ v16bf frag_A_T(const bf16* src, int ld){ // A[m][k] = src[k*ld+m]
  int lane = threadIdx.x & 31; int m = lane & 15; int hi = lane >> 4;
  v16bf f;
#pragma unroll
  for(int e=0;e<16;e++) f[e] = src[kmap(e,hi)*ld + m];
  return f;
}
__device__ __forceinline__ v16bf frag_B(const bf16* src, int ld){ // B[k][n] = src[k*ld+n]
  int lane = threadIdx.x & 31; int n = lane & 15; int hi = lane >> 4;
  v16bf f;
#pragma unroll
  for(int e=0;e<16;e++) f[e] = src[kmap(e,hi)*ld + n];
  return f;
}
__device__ __forceinline__ v16bf frag_B_T(const bf16* src, int ld){ // B[k][n] = src[n*ld+k]
  int lane = threadIdx.x & 31; int n = lane & 15; int hi = lane >> 4;
  v16bf f;
#pragma unroll
  for(int e=0;e<16;e++) f[e] = src[n*ld + kmap(e,hi)];
  return f;
}

// ---------------------------------------------------------------- workgroup 64x64x64 GEMM
// D = alpha*(op(A) @ op(B)) + beta*Cbf ; all operands 64x64 (ld=64) bf16 in LDS.
// 8 waves, each computes 2 of the 16 (16x16) output tiles; 2 wmma per tile.
template<int TA, int TB>
__device__ void wg_mm64(const bf16* A, const bf16* B, const bf16* Cbf,
                        float alpha, float beta,
                        bf16* Dbf, float* Df, int ldD, float* red){
  __syncthreads();
  const int tid  = threadIdx.x;
  const int wave = tid >> 5;
  const int lane = tid & 31;
  const int n_   = lane & 15;
  const int mo   = (lane >> 4) << 3;
  float sq = 0.f;
#pragma unroll
  for(int t = wave; t < 16; t += 8){
    const int mi = t >> 2, ni = t & 3;
    v8f acc = {0,0,0,0,0,0,0,0};
#pragma unroll
    for(int kk=0; kk<64; kk+=32){
      v16bf a = TA ? frag_A_T(A + kk*64 + mi*16, 64)
                   : frag_A  (A + mi*16*64 + kk, 64);
      v16bf b = TB ? frag_B_T(B + ni*16*64 + kk, 64)
                   : frag_B  (B + kk*64 + ni*16, 64);
      acc = wmma_bf16(a, b, acc);
    }
#pragma unroll
    for(int r=0;r<8;r++){
      float v = alpha * acc[r];
      if(beta != 0.f) v += beta * (float)Cbf[(mi*16+mo+r)*64 + ni*16 + n_];
      if(Dbf) Dbf[(mi*16+mo+r)*64 + ni*16 + n_] = (bf16)v;
      if(Df)  Df [(size_t)(mi*16+mo+r)*ldD + ni*16 + n_] = v;
      sq += v*v;
    }
  }
  if(red) red[tid] = sq;
  __syncthreads();
}

// ---------------------------------------------------------------- Newton-Schulz (5 iters, bf16)
__device__ bf16* ns5(bf16* Xa, bf16* Xb, bf16* Ab, bf16* Bb, float* red){
  const int tid = threadIdx.x;
  for(int s=128; s>0; s>>=1){ if(tid<s) red[tid]+=red[tid+s]; __syncthreads(); }
  const float inv = 1.0f/(sqrtf(red[0]) + 1e-7f);
  for(int u=tid; u<4096; u+=256) Xa[u] = (bf16)((float)Xa[u]*inv);
  bf16* Xc = Xa; bf16* Xn = Xb;
  for(int it=0; it<5; ++it){
    wg_mm64<0,1>(Xc, Xc, nullptr, 1.0f,  0.0f, Ab, nullptr, 0, nullptr); // A = X X^T
    wg_mm64<0,0>(Ab, Ab, Ab,      C_NSC, B_NSC, Bb, nullptr, 0, nullptr); // B = b*A + c*A*A
    wg_mm64<0,0>(Bb, Xc, Xc,      1.0f,  A_NSC, Xn, nullptr, 0, nullptr); // X = a*X + B*X
    bf16* tp = Xc; Xc = Xn; Xn = tp;
  }
  return Xc;
}

// cw += g ; cw = cw / (||col||+1e-5) * wn ; master f32 in global (L2-resident), bf16 in LDS.
__device__ void update_w(float* Wg, const bf16* g, bf16* sw, const float* wn,
                         float* red, int tid){
  const int j = tid & 63, gp = tid >> 6, r0 = gp*16;
  float p = 0.f;
  for(int r=r0; r<r0+16; ++r){ int u=r*64+j; float v=Wg[u]+(float)g[u]; Wg[u]=v; p+=v*v; }
  red[tid]=p; __syncthreads();
  if(tid<64) red[tid] = sqrtf(red[tid]+red[tid+64]+red[tid+128]+red[tid+192]);
  __syncthreads();
  const float sc = wn[j] / (red[j] + 1e-5f);
  for(int r=r0; r<r0+16; ++r){ int u=r*64+j; float v=Wg[u]*sc; Wg[u]=v; sw[u]=(bf16)v; }
  __syncthreads();
}

__device__ void init_w(float* Wg, const float* src, bf16* sw, float* wn,
                       float* red, int tid){
  const int j = tid & 63, gp = tid >> 6, r0 = gp*16;
  float p = 0.f;
  for(int r=r0; r<r0+16; ++r){ int u=r*64+j; float v=src[u]; Wg[u]=v; sw[u]=(bf16)v; p+=v*v; }
  red[tid]=p; __syncthreads();
  if(tid<64) wn[tid] = sqrtf(red[tid]+red[tid+64]+red[tid+128]+red[tid+192]);
  __syncthreads();
}

// ---------------------------------------------------------------- kernels
__global__ __launch_bounds__(256) void k_cvt(const float* src, bf16* dst, int n){
  int i = blockIdx.x*256 + threadIdx.x;
  if(i < n) dst[i] = (bf16)src[i];
}

// per-token learning rates: softplus(x @ Wlr^T + blr + BASE_LR_INV) -> lrw[which][bh][L]
__global__ __launch_bounds__(256) void k_lr(const float* x, const float* Wlr,
                                            const float* blr, float* lrw){
  __shared__ float sx[1024];
  const int row = blockIdx.x;                 // 0..16383 (b*L)
  const float* xr = x + (size_t)row*1024;
  for(int u=threadIdx.x; u<1024; u+=256) sx[u] = xr[u];
  __syncthreads();
  const int j = threadIdx.x;
  if(j < 48){
    const float* wr = Wlr + (size_t)j*1024;
    float acc = 0.f;
    for(int k=0;k<1024;k++) acc += sx[k]*wr[k];
    acc += blr[j] + BASE_LR_INV_F;
    float sp = (acc > 20.f) ? acc : log1pf(__expf(acc));
    const int which = j>>4, head = j&15;
    const int bi = row>>12, l = row&4095;
    lrw[ ((size_t)(which*64 + bi*16 + head))*4096 + l ] = sp;
  }
}

// qkv = silu(x @ Wqkv^T), split heads, L2-normalize q,k over hd; scatter [bh][L][hd] bf16
// K=64 stages, LDS double-buffered, async global->LDS; 1 barrier per 4 wmma/wave.
__global__ __launch_bounds__(256) void k_qkv(const bf16* xbf, const bf16* wbf,
                                             bf16* qb, bf16* kb, bf16* vb){
  __shared__ bf16 sA[2][4096];
  __shared__ bf16 sB[2][4096];
  __shared__ float sC[4096];
  __shared__ float srn[64];
  const int tid=threadIdx.x, wave=tid>>5, lane=tid&31;
  const int m0 = blockIdx.x*64;
  const int ntile = blockIdx.y, n0 = ntile*64;
  const int u0 = tid*16, r_ = u0>>6, c_ = u0&63;      // 16 elems staged per thread per buffer
  const bf16* Asrc = xbf + (size_t)(m0+r_)*1024 + c_;
  const bf16* Bsrc = wbf + (size_t)(n0+r_)*1024 + c_;
  v8f acc0 = {0,0,0,0,0,0,0,0};
  v8f acc1 = {0,0,0,0,0,0,0,0};
  const int mi0=wave>>2, ni0=wave&3, mi1=(wave+8)>>2, ni1=(wave+8)&3;
  // prologue: stage K-block 0 into buf 0
  cp16_async(Asrc,     &sA[0][u0]);  cp16_async(Asrc + 8, &sA[0][u0+8]);
  cp16_async(Bsrc,     &sB[0][u0]);  cp16_async(Bsrc + 8, &sB[0][u0+8]);
  for(int it=0; it<16; ++it){
    const int buf = it & 1;
    async_wait0();
    __syncthreads();
    if(it+1 < 16){
      const int k0 = (it+1)*64;
      cp16_async(Asrc + k0,     &sA[buf^1][u0]);
      cp16_async(Asrc + k0 + 8, &sA[buf^1][u0+8]);
      cp16_async(Bsrc + k0,     &sB[buf^1][u0]);
      cp16_async(Bsrc + k0 + 8, &sB[buf^1][u0+8]);
    }
    const bf16* A = sA[buf]; const bf16* Bm = sB[buf];
#pragma unroll
    for(int kk=0; kk<64; kk+=32){
      acc0 = wmma_bf16(frag_A(A+mi0*16*64+kk,64), frag_B_T(Bm+ni0*16*64+kk,64), acc0);
      acc1 = wmma_bf16(frag_A(A+mi1*16*64+kk,64), frag_B_T(Bm+ni1*16*64+kk,64), acc1);
    }
  }
  const int n_=lane&15, mo=(lane>>4)<<3;
#pragma unroll
  for(int r=0;r<8;r++) sC[(mi0*16+mo+r)*64 + ni0*16+n_] = acc0[r];
#pragma unroll
  for(int r=0;r<8;r++) sC[(mi1*16+mo+r)*64 + ni1*16+n_] = acc1[r];
  __syncthreads();
  for(int u=tid; u<4096; u+=256) sC[u] = siluf(sC[u]);
  __syncthreads();
  const int which = ntile>>4, head = ntile&15;
  if(which < 2 && tid < 64){
    float s=0.f;
    for(int e=0;e<64;e++){ float v=sC[tid*64+e]; s+=v*v; }
    srn[tid] = 1.0f/(sqrtf(s)+1e-5f);
  }
  __syncthreads();
  bf16* dst = (which==0)?qb:((which==1)?kb:vb);
  for(int u=tid; u<4096; u+=256){
    int r=u>>6, e=u&63;
    float v = sC[u];
    if(which<2) v *= srn[r];
    int gm=m0+r, bi=gm>>12, l=gm&4095;
    dst[ ((size_t)(bi*16+head)*4096 + l)*64 + e ] = (bf16)v;
  }
}

// ---- the TTT scan: 1 workgroup per (b,h), sequential over 64 chunks, all WMMA ----
__global__ __launch_bounds__(256) void k_scan(const bf16* qb, const bf16* kb, const bf16* vb,
                                              const float* lrw,
                                              const float* w0i, const float* w1i, const float* w2i,
                                              float* cwg, float* oi){
  __shared__ bf16 s_w0[4096], s_w1[4096], s_w2[4096];     // fast weights (bf16 working copies)
  __shared__ bf16 s_ki[4096], s_vi[4096];                  // chunk k,v (qi reuses s_ki later)
  __shared__ bf16 s_ga[4096], s_hp[4096];                  // gate, hid_pre (also NS X dbl-buf)
  __shared__ bf16 s_t0[4096], s_t1[4096], s_t2[4096], s_t3[4096]; // hidden*l1, dhid, dhid_pre, dgate_pre
  __shared__ bf16 s_k0[4096], s_k2[4096];                  // ki*l0, ki*l2
  __shared__ float s_red[256];
  __shared__ float s_lr[3][64];
  __shared__ float s_wn[3][64];

  const int tid  = threadIdx.x;
  const int bhI  = blockIdx.x;          // 0..63
  const int head = bhI & 15;
  float* W0 = cwg + (size_t)(  0 + bhI)*4096;
  float* W1 = cwg + (size_t)( 64 + bhI)*4096;
  float* W2 = cwg + (size_t)(128 + bhI)*4096;

  init_w(W0, w0i + (size_t)head*4096, s_w0, s_wn[0], s_red, tid);
  init_w(W1, w1i + (size_t)head*4096, s_w1, s_wn[1], s_red, tid);
  init_w(W2, w2i + (size_t)head*4096, s_w2, s_wn[2], s_red, tid);

  const int u0 = tid*16;                // 16 elems per thread per 64x64 tile
  for(int c=0; c<64; ++c){
    const size_t coff = ((size_t)bhI*4096 + c*64)*64;
    // stage ki, vi (async direct-to-LDS), lr
    cp16_async(kb + coff + u0,     s_ki + u0);
    cp16_async(kb + coff + u0 + 8, s_ki + u0 + 8);
    cp16_async(vb + coff + u0,     s_vi + u0);
    cp16_async(vb + coff + u0 + 8, s_vi + u0 + 8);
    if(tid < 192){
      int m = tid>>6, i = tid&63;
      s_lr[m][i] = lrw[ ((size_t)(m*64 + bhI))*4096 + c*64 + i ];
    }
    async_wait0();
    // forward: gate = ki@w0 ; hid_pre = ki@w2
    wg_mm64<0,0>(s_ki, s_w0, nullptr, 1.f, 0.f, s_ga, nullptr, 0, nullptr);
    wg_mm64<0,0>(s_ki, s_w2, nullptr, 1.f, 0.f, s_hp, nullptr, 0, nullptr);
    // hidden*l1, ki*l0, ki*l2
    for(int u=tid; u<4096; u+=256){
      int r = u>>6;
      float g=(float)s_ga[u], hp=(float)s_hp[u], kv=(float)s_ki[u];
      s_t0[u] = (bf16)(siluf(g)*hp*s_lr[1][r]);
      s_k0[u] = (bf16)(kv*s_lr[0][r]);
      s_k2[u] = (bf16)(kv*s_lr[2][r]);
    }
    // dhid = vi @ w1^T
    wg_mm64<0,1>(s_vi, s_w1, nullptr, 1.f, 0.f, s_t1, nullptr, 0, nullptr);
    // dhid_pre = dhid*silu(gate) ; dgate_pre = silu_bp(dhid*hid_pre, gate)
    for(int u=tid; u<4096; u+=256){
      float g=(float)s_ga[u], hp=(float)s_hp[u], dh=(float)s_t1[u];
      s_t2[u] = (bf16)(dh*siluf(g));
      s_t3[u] = (bf16)silu_bp(dh*hp, g);
    }
    // w1 grad: (hidden*l1)^T @ vi -> NS -> update   (s_ga/s_hp/s_ki/s_vi free as NS scratch)
    wg_mm64<1,0>(s_t0, s_vi, nullptr, 1.f, 0.f, s_ga, nullptr, 0, s_red);
    update_w(W1, ns5(s_ga, s_hp, s_ki, s_vi, s_red), s_w1, s_wn[1], s_red, tid);
    // w0 grad: (ki*l0)^T @ dgate_pre
    wg_mm64<1,0>(s_k0, s_t3, nullptr, 1.f, 0.f, s_ga, nullptr, 0, s_red);
    update_w(W0, ns5(s_ga, s_hp, s_ki, s_vi, s_red), s_w0, s_wn[0], s_red, tid);
    // w2 grad: (ki*l2)^T @ dhid_pre
    wg_mm64<1,0>(s_k2, s_t2, nullptr, 1.f, 0.f, s_ga, nullptr, 0, s_red);
    update_w(W2, ns5(s_ga, s_hp, s_ki, s_vi, s_red), s_w2, s_wn[2], s_red, tid);
    // output: oi = (silu(qi@w0) * (qi@w2)) @ w1  (updated weights)
    cp16_async(qb + coff + u0,     s_ki + u0);
    cp16_async(qb + coff + u0 + 8, s_ki + u0 + 8);
    async_wait0();
    wg_mm64<0,0>(s_ki, s_w0, nullptr, 1.f, 0.f, s_ga, nullptr, 0, nullptr);
    wg_mm64<0,0>(s_ki, s_w2, nullptr, 1.f, 0.f, s_hp, nullptr, 0, nullptr);
    for(int u=tid; u<4096; u+=256)
      s_vi[u] = (bf16)(siluf((float)s_ga[u]) * (float)s_hp[u]);
    wg_mm64<0,0>(s_vi, s_w1, nullptr, 1.f, 0.f, nullptr, oi + coff, 64, nullptr);
  }
}

// RMSNorm over hd, scale by o_norm_w, scatter to [b][L][h*hd] bf16
__global__ __launch_bounds__(256) void k_rms(const float* oi, const float* onw, bf16* obf){
  const int gid  = blockIdx.x*256 + threadIdx.x;
  const int w    = gid >> 5;            // token index within [bh*L]
  const int lane = gid & 31;
  const int bhI = w >> 12, l = w & 4095;
  const float* row = oi + (size_t)w*64;
  float v0 = row[lane], v1 = row[lane+32];
  float s = v0*v0 + v1*v1;
#pragma unroll
  for(int m=16; m>0; m>>=1) s += __shfl_xor(s, m, 32);
  const float r = rsqrtf(s*(1.0f/64.0f) + 1e-5f);
  const int bi = bhI >> 4, head = bhI & 15;
  bf16* drow = obf + ((size_t)(bi*4096 + l))*1024 + head*64;
  drow[lane]    = (bf16)(v0*r*onw[lane]);
  drow[lane+32] = (bf16)(v1*r*onw[lane+32]);
}

// out = obf @ Wproj^T  (f32 output); same double-buffered async mainloop
__global__ __launch_bounds__(256) void k_proj(const bf16* obf, const bf16* wbf, float* out){
  __shared__ bf16 sA[2][4096];
  __shared__ bf16 sB[2][4096];
  const int tid=threadIdx.x, wave=tid>>5, lane=tid&31;
  const int m0 = blockIdx.x*64, n0 = blockIdx.y*64;
  const int u0 = tid*16, r_ = u0>>6, c_ = u0&63;
  const bf16* Asrc = obf + (size_t)(m0+r_)*1024 + c_;
  const bf16* Bsrc = wbf + (size_t)(n0+r_)*1024 + c_;
  v8f acc0 = {0,0,0,0,0,0,0,0};
  v8f acc1 = {0,0,0,0,0,0,0,0};
  const int mi0=wave>>2, ni0=wave&3, mi1=(wave+8)>>2, ni1=(wave+8)&3;
  cp16_async(Asrc,     &sA[0][u0]);  cp16_async(Asrc + 8, &sA[0][u0+8]);
  cp16_async(Bsrc,     &sB[0][u0]);  cp16_async(Bsrc + 8, &sB[0][u0+8]);
  for(int it=0; it<16; ++it){
    const int buf = it & 1;
    async_wait0();
    __syncthreads();
    if(it+1 < 16){
      const int k0 = (it+1)*64;
      cp16_async(Asrc + k0,     &sA[buf^1][u0]);
      cp16_async(Asrc + k0 + 8, &sA[buf^1][u0+8]);
      cp16_async(Bsrc + k0,     &sB[buf^1][u0]);
      cp16_async(Bsrc + k0 + 8, &sB[buf^1][u0+8]);
    }
    const bf16* A = sA[buf]; const bf16* Bm = sB[buf];
#pragma unroll
    for(int kk=0; kk<64; kk+=32){
      acc0 = wmma_bf16(frag_A(A+mi0*16*64+kk,64), frag_B_T(Bm+ni0*16*64+kk,64), acc0);
      acc1 = wmma_bf16(frag_A(A+mi1*16*64+kk,64), frag_B_T(Bm+ni1*16*64+kk,64), acc1);
    }
  }
  const int n_=lane&15, mo=(lane>>4)<<3;
#pragma unroll
  for(int r=0;r<8;r++) out[(size_t)(m0+mi0*16+mo+r)*1024 + n0+ni0*16+n_] = acc0[r];
#pragma unroll
  for(int r=0;r<8;r++) out[(size_t)(m0+mi1*16+mo+r)*1024 + n0+ni1*16+n_] = acc1[r];
}

// ---------------------------------------------------------------- launch
extern "C" void kernel_launch(void* const* d_in, const int* in_sizes, int n_in,
                              void* d_out, int out_size, void* d_ws, size_t ws_size,
                              hipStream_t stream) {
  const float* x     = (const float*)d_in[0];
  const float* Wqkv  = (const float*)d_in[1];
  const float* Wproj = (const float*)d_in[2];
  const float* Wlr   = (const float*)d_in[3];
  const float* blr   = (const float*)d_in[4];
  const float* w0    = (const float*)d_in[5];
  const float* w1    = (const float*)d_in[6];
  const float* w2    = (const float*)d_in[7];
  const float* onw   = (const float*)d_in[8];
  float* out = (float*)d_out;

  char* p = (char*)d_ws;
  bf16* xbf   = (bf16*)p;  p += (size_t)16384*1024*2;
  bf16* wqkvb = (bf16*)p;  p += (size_t)3072*1024*2;
  bf16* wprjb = (bf16*)p;  p += (size_t)1024*1024*2;
  bf16* qb    = (bf16*)p;  p += (size_t)64*4096*64*2;
  bf16* kb    = (bf16*)p;  p += (size_t)64*4096*64*2;
  bf16* vb    = (bf16*)p;  p += (size_t)64*4096*64*2;
  float* lrw  = (float*)p; p += (size_t)3*64*4096*4;
  float* cwg  = (float*)p; p += (size_t)3*64*4096*4;
  float* oi   = (float*)p; p += (size_t)64*4096*64*4;
  bf16* obf   = (bf16*)p;  p += (size_t)16384*1024*2;

  k_cvt<<<(16384*1024)/256, 256, 0, stream>>>(x, xbf, 16384*1024);
  k_cvt<<<(3072*1024)/256,  256, 0, stream>>>(Wqkv, wqkvb, 3072*1024);
  k_cvt<<<(1024*1024)/256,  256, 0, stream>>>(Wproj, wprjb, 1024*1024);
  k_lr<<<16384, 256, 0, stream>>>(x, Wlr, blr, lrw);
  k_qkv<<<dim3(256,48), 256, 0, stream>>>(xbf, wqkvb, qb, kb, vb);
  k_scan<<<64, 256, 0, stream>>>(qb, kb, vb, lrw, w0, w1, w2, cwg, oi);
  k_rms<<<(64*4096*32)/256, 256, 0, stream>>>(oi, onw, obf);
  k_proj<<<dim3(256,16), 256, 0, stream>>>(obf, wprjb, out);
}
